// Bezier_44083544326680
// MI455X (gfx1250) — matrix operand order repaired and checked by reading, not compiled, not run on hardware
//
#include <hip/hip_runtime.h>

// ---------------------------------------------------------------------------
// Bezier Gaussian rasterizer for MI455X (gfx1250, wave32, WMMA).
//
// patch[s] = gx[s] (x) gy[s]  (separable Gaussian)  =>
// sum over a group of K=64 consecutive steps (curve moves <1px/step):
//     patch32x32 = Gx(32x64) * Gy(64x32)
// computed as 2 chained K=32 slabs of 4x v_wmma_f32_16x16x32_f16,
// then ONE 32x32 atomic scatter per wave (64x fewer atomics than naive).
// Output (16.8 MB) is L2-resident on the 192 MB L2.
// ---------------------------------------------------------------------------

#define RES    2048
#define STEPS  65536
#define WIN    32
#define CHUNK  64                  // steps per wave (2 WMMA K-batches of 32)
#define KB     (CHUNK / 32)
#define WPB    4                   // waves per block
#define BLOCK  (WPB * 32)

typedef __attribute__((ext_vector_type(16))) _Float16 v16h;
typedef __attribute__((ext_vector_type(8)))  float    v8f;

__global__ __launch_bounds__(BLOCK)
void bezier_wmma_raster(const float* __restrict__ cp, float* __restrict__ out)
{
    const float inv2s2 = 1.0f / (2.0f * 0.01f * 0.01f);   // 1/(2*sigma^2)
    const float invRES = 1.0f / (float)RES;
    const float scale  = 1.0f / (float)STEPS;

    const int lane = threadIdx.x & 31;
    const int wv   = threadIdx.x >> 5;
    const int wave = blockIdx.x * WPB + wv;
    const int s0   = wave * CHUNK;

    // 3 control points (x,y)
    const float p0x = cp[0], p0y = cp[1];
    const float p1x = cp[2], p1y = cp[3];
    const float p2x = cp[4], p2y = cp[5];

    __shared__ float s_cx[WPB][CHUNK];
    __shared__ float s_cy[WPB][CHUNK];

    // --- evaluate curve samples for this wave's CHUNK steps; stage in LDS ---
    int bxm = RES, bym = RES;
    for (int b = 0; b < KB; ++b) {
        const int   s  = s0 + b * 32 + lane;
        const float tl = (float)s * (1.0f / (float)(STEPS - 1)); // linspace
        const float to = (float)s * (1.0f / (float)STEPS);       // arange/steps
        const float ax = p0x + (p1x - p0x) * tl;
        const float ay = p0y + (p1y - p0y) * tl;
        const float bx = p1x + (p2x - p1x) * tl;
        const float by = p1y + (p2y - p1y) * tl;
        const float cx = ax + to * (bx - ax);
        const float cy = ay + to * (by - ay);
        s_cx[wv][b * 32 + lane] = cx;
        s_cy[wv][b * 32 + lane] = cy;
        int bxi = min(max((int)floorf((float)RES * cx) - WIN / 2, 0), RES - WIN);
        int byi = min(max((int)floorf((float)RES * cy) - WIN / 2, 0), RES - WIN);
        bxm = min(bxm, bxi);
        bym = min(bym, byi);
    }
    // wave-wide min => common 32x32 window corner for the whole group
    for (int off = 16; off > 0; off >>= 1) {
        bxm = min(bxm, __shfl_xor(bxm, off, 32));
        bym = min(bym, __shfl_xor(bym, off, 32));
    }
    __syncthreads();

    // --- accumulate sum of outer products as a 32x64 * 64x32 GEMM ----------
    v8f d00 = {}, d01 = {}, d10 = {}, d11 = {};

    const int halfsel = (lane & 16) >> 1;   // A-fragment K offset (0 or 8)
    const int mrow    = lane & 15;          // row/col index within 16-tile

    for (int b = 0; b < KB; ++b) {
        v16h a0 = {}, a1 = {}, b0 = {}, b1 = {};

        // A (16-bit 16x32 layout): lane<16 holds K={0..7,16..23}, lane>=16 K+8
        #pragma unroll
        for (int e = 0; e < 16; ++e) {
            const int   k   = b * 32 + e + (e & 8) + halfsel;
            const float cxk = s_cx[wv][k];
            const float x0  = (float)(bxm + mrow)      * invRES;  // tile r=0
            const float x1  = (float)(bxm + mrow + 16) * invRES;  // tile r=1
            const float dx0 = cxk - x0;
            const float dx1 = cxk - x1;
            a0[e] = (_Float16)__expf(-dx0 * dx0 * inv2s2);
            a1[e] = (_Float16)__expf(-dx1 * dx1 * inv2s2);
        }
        // B (32x16): N=lane&15; lane<16 holds K={0..7,16..23}, lane>=16 K+8
        #pragma unroll
        for (int e = 0; e < 16; ++e) {
            const int   k   = b * 32 + e + (e & 8) + halfsel;
            const float cyk = s_cy[wv][k];
            const float y0  = (float)(bym + mrow)      * invRES;  // tile c=0
            const float y1  = (float)(bym + mrow + 16) * invRES;  // tile c=1
            const float dy0 = cyk - y0;
            const float dy1 = cyk - y1;
            b0[e] = (_Float16)__expf(-dy0 * dy0 * inv2s2);
            b1[e] = (_Float16)__expf(-dy1 * dy1 * inv2s2);
        }

        // 4 tiles of 16x16, f32 accumulate chained across K-batches
        d00 = __builtin_amdgcn_wmma_f32_16x16x32_f16(false, a0, false, b0,
                                                     (short)0, d00, false, false);
        d01 = __builtin_amdgcn_wmma_f32_16x16x32_f16(false, a0, false, b1,
                                                     (short)0, d01, false, false);
        d10 = __builtin_amdgcn_wmma_f32_16x16x32_f16(false, a1, false, b0,
                                                     (short)0, d10, false, false);
        d11 = __builtin_amdgcn_wmma_f32_16x16x32_f16(false, a1, false, b1,
                                                     (short)0, d11, false, false);
    }

    // --- one 32x32 atomic scatter per wave (L2-resident raster) ------------
    // D layout: VGPR v -> M = v (+8 for lanes 16-31), N = lane&15
    const int rbase = (lane < 16) ? 0 : 8;
    const int col0  = bym + mrow;
    #pragma unroll
    for (int v = 0; v < 8; ++v) {
        const int r0 = bxm + rbase + v;
        unsafeAtomicAdd(&out[(r0     ) * RES + col0     ], d00[v] * scale);
        unsafeAtomicAdd(&out[(r0     ) * RES + col0 + 16], d01[v] * scale);
        unsafeAtomicAdd(&out[(r0 + 16) * RES + col0     ], d10[v] * scale);
        unsafeAtomicAdd(&out[(r0 + 16) * RES + col0 + 16], d11[v] * scale);
    }
}

extern "C" void kernel_launch(void* const* d_in, const int* in_sizes, int n_in,
                              void* d_out, int out_size, void* d_ws, size_t ws_size,
                              hipStream_t stream)
{
    const float* cp  = (const float*)d_in[0];   // control_points: 6 floats
    float*       out = (float*)d_out;           // 2048*2048 floats

    // harness poisons d_out; raster accumulates, so zero it (capture-legal)
    hipMemsetAsync(out, 0, (size_t)out_size * sizeof(float), stream);

    const int waves  = STEPS / CHUNK;           // 1024 waves
    const int blocks = waves / WPB;             // 256 blocks of 4 waves
    bezier_wmma_raster<<<blocks, BLOCK, 0, stream>>>(cp, out);
}
